// AttentionTFIDF_26104811225625
// MI455X (gfx1250) — matrix operand order repaired and checked by reading, not compiled, not run on hardware
//
#include <hip/hip_runtime.h>
#include <hip/hip_bf16.h>

typedef __attribute__((ext_vector_type(16))) __bf16 bf16x16;
typedef __attribute__((ext_vector_type(8)))  float  f32x8;

constexpr int kB  = 256;   // docs (grid)
constexpr int kL  = 256;   // tokens per doc
constexpr int kH  = 256;   // hidden
constexpr int kC  = 20;    // classes
constexpr int kPH = 264;   // padded LDS row pitch (elements) -> 528B rows, 16B aligned chunks

union Frag16 { bf16x16 v; uint4 q[2]; };
union Pack8  { __bf16 h[8]; uint4 q; };

__global__ __launch_bounds__(256, 1)
void attn_tfidf_kernel(const int* __restrict__ doc_tids,
                       const int* __restrict__ TFs,
                       const int* __restrict__ DFs,
                       const float* __restrict__ emb,
                       const float* __restrict__ fc_w,
                       const float* __restrict__ fc_b,
                       float* __restrict__ out)
{
    extern __shared__ char smem[];
    __bf16*   hb = (__bf16*)smem;                         // [kL][kPH] bf16 embeddings
    _Float16* Eh = (_Float16*)(smem + kL * kPH * 2);      // [kL][kPH] f16: exp(co)*tfidf_j
    float* fbase  = (float*)(smem + 2 * kL * kPH * 2);
    float* s2     = fbase;          // [256] ||h_i||^2 (of bf16-rounded h)
    float* tfidf  = fbase + 256;    // [256]
    float* validf = fbase + 512;    // [256] 1.0 valid / 0.0 pad
    float* rowCo  = fbase + 768;    // [256] sum_j co_ij
    float* rowExp = fbase + 1024;   // [256] sum_j exp(co_ij)
    float* alpha  = fbase + 1280;   // [256] weights_i / rowExp_i
    float* gcol   = fbase + 1536;   // [256] sum_i alpha_i * E'_ij
    float* finalv = fbase + 1792;   // [256] pooled doc vector
    float* partA  = fbase + 2048;   // [8]
    float* partB  = fbase + 2056;   // [8]
    float* lgbuf  = fbase + 2064;   // [20]
    float* scal   = fbase + 2084;   // [1] doc_size

    const int t    = threadIdx.x;
    const int w    = t >> 5;        // wave id (8 waves, wave32)
    const int l    = t & 31;        // lane
    const int half = l >> 4;
    const int lr   = l & 15;
    const int b    = blockIdx.x;

    // ---------------- Phase 1: tfidf / valid / gather h -> LDS bf16, s2 ----------------
    {
        int  tid = doc_tids[b * kL + t];
        float tf = fminf(fmaxf((float)TFs[b * kL + t], 1.f), 20.f);
        float idf = __logf(100001.f / ((float)DFs[b * kL + t] + 1.f));
        tfidf[t]  = (1.f + __logf(tf)) * idf;
        float vv  = (tid != 0) ? 1.f : 0.f;
        validf[t] = vv;
        float s = vv;
        #pragma unroll
        for (int m = 16; m >= 1; m >>= 1) s += __shfl_xor(s, m, 32);
        if (l == 0) partA[w] = s;
    }
    for (int rr = 0; rr < 32; ++rr) {
        int r   = w * 32 + rr;
        int tok = doc_tids[b * kL + r];
        const float4* src = (const float4*)(emb + (size_t)tok * kH + l * 8);
        float4 x0 = src[0];
        float4 x1 = src[1];
        Pack8 u;
        u.h[0] = (__bf16)x0.x; u.h[1] = (__bf16)x0.y; u.h[2] = (__bf16)x0.z; u.h[3] = (__bf16)x0.w;
        u.h[4] = (__bf16)x1.x; u.h[5] = (__bf16)x1.y; u.h[6] = (__bf16)x1.z; u.h[7] = (__bf16)x1.w;
        *(uint4*)(hb + r * kPH + l * 8) = u.q;
        float sq = 0.f;
        #pragma unroll
        for (int k = 0; k < 8; ++k) { float xb = (float)u.h[k]; sq += xb * xb; }
        #pragma unroll
        for (int m = 16; m >= 1; m >>= 1) sq += __shfl_xor(sq, m, 32);
        if (l == 0) s2[r] = sq;
    }
    __syncthreads();
    if (t == 0) {
        float s = 0.f;
        for (int k = 0; k < 8; ++k) s += partA[k];
        scal[0] = s;
    }
    __syncthreads();

    // ---------------- Phase 2: gram matrix via WMMA + fused distance/softmax epilogue ----
    {
        Frag16 a[8];
        for (int it = w * 2; it <= w * 2 + 1; ++it) {
            // A fragments for this 16-row block (resident across all j tiles)
            const __bf16* arow = hb + (it * 16 + lr) * kPH;
            #pragma unroll
            for (int ks = 0; ks < 8; ++ks) {
                const __bf16* p = arow + ks * 32 + half * 8;
                a[ks].q[0] = *(const uint4*)(p);
                a[ks].q[1] = *(const uint4*)(p + 16);
            }
            float accCo[8], accExp[8];
            #pragma unroll
            for (int v = 0; v < 8; ++v) { accCo[v] = 0.f; accExp[v] = 0.f; }

            for (int jt = 0; jt < 16; ++jt) {
                f32x8 c = {0.f, 0.f, 0.f, 0.f, 0.f, 0.f, 0.f, 0.f};
                const __bf16* brow = hb + (jt * 16 + lr) * kPH + half * 8;
                #pragma unroll
                for (int ks = 0; ks < 8; ++ks) {
                    Frag16 bf;
                    const __bf16* q = brow + ks * 32;
                    bf.q[0] = *(const uint4*)(q);
                    bf.q[1] = *(const uint4*)(q + 16);
                    c = __builtin_amdgcn_wmma_f32_16x16x32_bf16(
                            false, a[ks].v, false, bf.v, (short)0, c, false, false);
                }
                #pragma unroll
                for (int v = 0; v < 8; ++v) {
                    int i = it * 16 + v + half * 8;
                    int j = jt * 16 + lr;
                    float d2 = s2[i] + s2[j] - 2.f * c[v];
                    d2 = fmaxf(d2, 0.f);
                    float co = -sqrtf(d2 + 1e-12f) * 0.0625f;   // /sqrt(H)=1/16
                    co *= validf[i] * validf[j];                 // masked -> 0
                    float e = __expf(co);
                    accCo[v]  += co;
                    accExp[v] += e;
                    Eh[i * kPH + j] = (_Float16)(e * tfidf[j]);
                }
            }
            #pragma unroll
            for (int v = 0; v < 8; ++v) {
                float sc = accCo[v], se = accExp[v];
                #pragma unroll
                for (int m = 8; m >= 1; m >>= 1) {
                    sc += __shfl_xor(sc, m, 32);
                    se += __shfl_xor(se, m, 32);
                }
                if (lr == 0) {
                    int i = it * 16 + v + half * 8;
                    rowCo[i]  = sc;
                    rowExp[i] = se;
                }
            }
        }
    }
    __syncthreads();

    // ---------------- Phase 3: token weights = softmax_i(rowCo/doc_size) ----------------
    {
        float ds = scal[0];
        float wt = (validf[t] > 0.f && ds > 0.f) ? (rowCo[t] / ds) : -__builtin_inff();
        float m = wt;
        #pragma unroll
        for (int s = 16; s >= 1; s >>= 1) m = fmaxf(m, __shfl_xor(m, s, 32));
        if (l == 0) partA[w] = m;
        __syncthreads();
        float bm = partA[0];
        for (int k = 1; k < 8; ++k) bm = fmaxf(bm, partA[k]);
        float ew = (ds > 0.f) ? __expf(wt - bm) : 0.f;
        float s = ew;
        #pragma unroll
        for (int k = 16; k >= 1; k >>= 1) s += __shfl_xor(s, k, 32);
        if (l == 0) partB[w] = s;
        __syncthreads();
        float bs = partB[0];
        for (int k = 1; k < 8; ++k) bs += partB[k];
        float wgt = (ds > 0.f && bs > 0.f) ? (ew / bs) : 0.f;   // NaN cleanup path
        alpha[t] = wgt / rowExp[t];
    }
    __syncthreads();

    // ---------------- Phase 4: g_j = sum_i alpha_i * E'_ij  (wT P diag(tfidf)) ----------
    {
        float gj = 0.f;
        for (int i = 0; i < kL; ++i) gj += alpha[i] * (float)Eh[i * kPH + t];
        gcol[t] = gj;
    }
    __syncthreads();

    // ---------------- Phase 5: final_h = sum_j g_j * h_jh ------------------------------
    {
        float fh = 0.f;
        for (int j = 0; j < kL; ++j) fh += gcol[j] * (float)hb[j * kPH + t];
        finalv[t] = fh;
    }
    __syncthreads();

    // ---------------- Phase 6: fc + softmax over 20 classes ----------------------------
    if (t < kC) {
        float lg = fc_b[t];
        for (int h = 0; h < kH; ++h) lg += finalv[h] * fc_w[h * kC + t];
        lgbuf[t] = lg;
    }
    __syncthreads();
    if (t < kC) {
        float m = lgbuf[0];
        for (int k = 1; k < kC; ++k) m = fmaxf(m, lgbuf[k]);
        float s = 0.f;
        for (int k = 0; k < kC; ++k) s += __expf(lgbuf[k] - m);
        out[b * kC + t] = __expf(lgbuf[t] - m) / s;
    }
}

extern "C" void kernel_launch(void* const* d_in, const int* in_sizes, int n_in,
                              void* d_out, int out_size, void* d_ws, size_t ws_size,
                              hipStream_t stream) {
    (void)in_sizes; (void)n_in; (void)d_ws; (void)ws_size; (void)out_size;
    const int*   doc_tids = (const int*)d_in[0];
    const int*   TFs      = (const int*)d_in[1];
    const int*   DFs      = (const int*)d_in[2];
    const float* emb      = (const float*)d_in[3];
    const float* fc_w     = (const float*)d_in[4];
    const float* fc_b     = (const float*)d_in[5];
    float* out = (float*)d_out;

    size_t shmem = (size_t)2 * kL * kPH * 2   // hb (bf16) + Eh (f16)
                 + (size_t)2112 * sizeof(float);
    // CDNA5 WGP has 320KB LDS; opt in to >64KB dynamic shared (idempotent, capture-safe).
    (void)hipFuncSetAttribute(reinterpret_cast<const void*>(&attn_tfidf_kernel),
                              hipFuncAttributeMaxDynamicSharedMemorySize, (int)shmem);
    attn_tfidf_kernel<<<kB, 256, shmem, stream>>>(doc_tids, TFs, DFs, emb, fc_w, fc_b, out);
}
